// ProbabilisticSelectiveWorldModel_73134703116523
// MI455X (gfx1250) — matrix operand order repaired and checked by reading, not compiled, not run on hardware
//
#include <hip/hip_runtime.h>
#include <hip/hip_bf16.h>
#include <math.h>

// ---------------------------------------------------------------------------
// CDNA5 (gfx1250) WMMA world-model forward pass, NHWC, 32x32 tile per wave.
// Inner loops: 8x global_load_b128 + 4x v_wmma_f32_16x16x32_f16 per k-step.
// ---------------------------------------------------------------------------

typedef __attribute__((ext_vector_type(16))) _Float16 v16h;
typedef __attribute__((ext_vector_type(8)))  _Float16 v8h;
typedef __attribute__((ext_vector_type(8)))  float    v8f;

#define ACT_NONE     0
#define ACT_RELU     1
#define ACT_SILU     2
#define ACT_SOFTPLUS 3

__device__ __forceinline__ float act_apply(float v, int act) {
  switch (act) {
    case ACT_RELU:     return v > 0.f ? v : 0.f;
    case ACT_SILU:     return v / (1.f + __expf(-v));
    case ACT_SOFTPLUS: return v > 20.f ? v : log1pf(__expf(v));
    default:           return v;
  }
}

__device__ __forceinline__ v8f wmma_f16(v16h a, v16h b, v8f c) {
  return __builtin_amdgcn_wmma_f32_16x16x32_f16(false, a, false, b, (short)0, c,
                                                false, false);
}

// Fragment element order (ISA 7.12.2):
//  A: element e <-> k = akb + ((e>>3)<<4) + (e&7)  => runs [0..7] and [16..23]
//  B: element e <-> k = bkb + ((e>>3)<<4) + (e&7)  => runs [0..7] and [16..23]
__device__ __forceinline__ v16h pack16(v8h lo, v8h hi) {
  v16h r;
#pragma unroll
  for (int i = 0; i < 8; ++i) { r[i] = lo[i]; r[i + 8] = hi[i]; }
  return r;
}

// ---------------------------------------------------------------------------
// Elementwise / data-rearrangement helpers
// ---------------------------------------------------------------------------

__global__ void k_f32_to_f16(const float* __restrict__ s, _Float16* __restrict__ d,
                             long n) {
  long i = (long)blockIdx.x * blockDim.x + threadIdx.x;
  if (i < n) d[i] = (_Float16)s[i];
}

// image: NCHW f32 (C=3) -> NHWC f16 padded to C=8
__global__ void k_img8(const float* __restrict__ x, _Float16* __restrict__ o) {
  long i = (long)blockIdx.x * blockDim.x + threadIdx.x;
  if (i >= 64L * 256 * 256 * 8) return;
  int c = i & 7;
  long p = i >> 3;
  int w = p & 255;
  long p2 = p >> 8;
  int h = p2 & 255;
  int b = (int)(p2 >> 8);
  float v = (c < 3) ? x[(((long)b * 3 + c) * 256 + h) * 256 + w] : 0.f;
  o[i] = (_Float16)v;
}

// conv weight OIHW -> [O][(kh*4+kw)*Ipad + ci] f16 (zero pad ci >= I)
__global__ void k_wr_conv(const float* __restrict__ w, _Float16* __restrict__ o,
                          int O, int I, int Ipad) {
  int i = blockIdx.x * blockDim.x + threadIdx.x;
  int tot = O * 16 * Ipad;
  if (i >= tot) return;
  int n = i / (16 * Ipad);
  int r = i - n * 16 * Ipad;
  int cell = r / Ipad, ci = r - cell * Ipad;
  int kh = cell >> 2, kw = cell & 3;
  float v = (ci < I) ? w[((n * I + ci) * 4 + kh) * 4 + kw] : 0.f;
  o[i] = (_Float16)v;
}

// deconv weight (I,O,4,4) -> [O][(kh*4+kw)*I + ci] with spatial flip baked in
__global__ void k_wr_deconv(const float* __restrict__ w, _Float16* __restrict__ o,
                            int I, int O) {
  int i = blockIdx.x * blockDim.x + threadIdx.x;
  int tot = O * 16 * I;
  if (i >= tot) return;
  int n = i / (16 * I);
  int r = i - n * 16 * I;
  int cell = r / I, ci = r - cell * I;
  int kh = cell >> 2, kw = cell & 3;
  o[i] = (_Float16)w[((ci * O + n) * 4 + (3 - kh)) * 4 + (3 - kw)];
}

// enc_fc weight: dst k = hw*256 + c  <- src k = c*256 + hw  (N=512, K=65536)
__global__ void k_wr_enc(const float* __restrict__ w, _Float16* __restrict__ o) {
  long i = (long)blockIdx.x * blockDim.x + threadIdx.x;
  if (i >= 512L * 65536) return;
  int n = (int)(i >> 16);
  int k = (int)(i & 65535);
  int hw = k >> 8, c = k & 255;
  o[i] = (_Float16)w[((long)n << 16) + (c << 8) + hw];
}

// dec_fc weight rows permuted: dst row n' = hw*128 + c <- src row n = c*256 + hw
__global__ void k_wr_dec(const float* __restrict__ w, _Float16* __restrict__ o) {
  long i = (long)blockIdx.x * blockDim.x + threadIdx.x;
  if (i >= 32768L * 128) return;
  int np = (int)(i >> 7);
  int k = (int)(i & 127);
  int hw = np >> 7, c = np & 127;
  int n = (c << 8) + hw;
  o[i] = (_Float16)w[((long)n << 7) + k];
}

__global__ void k_wr_decb(const float* __restrict__ b, float* __restrict__ o) {
  int i = blockIdx.x * blockDim.x + threadIdx.x;
  if (i >= 32768) return;
  int hw = i >> 7, c = i & 127;
  o[i] = b[(c << 8) + hw];
}

// generic zero-padded f32->f16 row copy: [N][K] -> [N][Kp]
__global__ void k_wr_pad(const float* __restrict__ w, _Float16* __restrict__ o,
                         int N, int K, int Kp) {
  int i = blockIdx.x * blockDim.x + threadIdx.x;
  if (i >= N * Kp) return;
  int n = i / Kp, k = i - n * Kp;
  o[i] = (_Float16)((k < K) ? w[(long)n * K + k] : 0.f);
}

// zcat: [64][Kp] = [z_prev(256) | action(6) | zeros]
__global__ void k_zcat(const float* __restrict__ z, const float* __restrict__ a,
                       _Float16* __restrict__ o, int Kp) {
  int i = blockIdx.x * blockDim.x + threadIdx.x;
  if (i >= 64 * Kp) return;
  int r = i / Kp, c = i - r * Kp;
  float v = (c < 256) ? z[r * 256 + c] : ((c < 262) ? a[r * 6 + (c - 256)] : 0.f);
  o[i] = (_Float16)v;
}

__global__ void k_concat2_h(const _Float16* __restrict__ a, int an,
                            const _Float16* __restrict__ b, int bn,
                            _Float16* __restrict__ dst, int rows) {
  int n = an + bn;
  int i = blockIdx.x * blockDim.x + threadIdx.x;
  if (i >= rows * n) return;
  int r = i / n, c = i - r * n;
  dst[i] = (c < an) ? a[(long)r * an + c] : b[(long)r * bn + (c - an)];
}

__global__ void k_split_halves(const float* __restrict__ src, int M, int C,
                               float* __restrict__ lo, float* __restrict__ hi) {
  int i = blockIdx.x * blockDim.x + threadIdx.x;
  if (i >= M * C) return;
  int r = i / C, c = i - r * C;
  lo[i] = src[(long)r * 2 * C + c];
  hi[i] = src[(long)r * 2 * C + C + c];
}

__global__ void k_reparam(const float* __restrict__ post, const float* __restrict__ eps,
                          float* __restrict__ z, float* __restrict__ mu,
                          float* __restrict__ lv, _Float16* __restrict__ ztgt,
                          int Bn, int L, int T) {
  int i = blockIdx.x * blockDim.x + threadIdx.x;
  if (i >= Bn * L) return;
  int b = i / L, c = i - b * L;
  float m = post[(long)b * 2 * L + c];
  float l = post[(long)b * 2 * L + L + c];
  float zz = m + eps[i] * __expf(0.5f * l);
  z[i] = zz; mu[i] = m; lv[i] = l;
  if (c < T) ztgt[(long)b * T + c] = (_Float16)zz;
}

__global__ void k_ssm(const float* __restrict__ x_in, const float* __restrict__ Delta,
                      const float* __restrict__ Bt, const float* __restrict__ Ct,
                      const float* __restrict__ A_log, const float* __restrict__ Dv,
                      const float* __restrict__ prev, float* __restrict__ state_out,
                      float* __restrict__ belF, _Float16* __restrict__ belH,
                      int Bn, int HIDn, int DS) {
  int i = blockIdx.x * blockDim.x + threadIdx.x;
  if (i >= Bn * HIDn) return;
  int b = i / HIDn, h = i - b * HIDn;
  float x = x_in[i];
  float dt = Delta[i];
  float bel = 0.f;
  for (int s = 0; s < DS; ++s) {
    float Aa = -__expf(A_log[h * DS + s]);
    float ab = __expf(dt * Aa);
    float st = ab * prev[(long)i * DS + s] + dt * Bt[b * DS + s] * x;
    state_out[(long)i * DS + s] = st;
    bel += st * Ct[b * DS + s];
  }
  bel += Dv[h] * x;
  belF[i] = bel;
  belH[i] = (_Float16)bel;
}

// ---------------------------------------------------------------------------
// Aligned WMMA GEMM, 32x32 tile per wave (2x2 subtiles, 4 WMMA / k-step):
//   out[M,N] = act(A[M,K] @ W[N,K]^T + bias)
// Requires: M % 32 == 0, K % 32 == 0. N edge handled via clamp + select.
// ---------------------------------------------------------------------------

__global__ __launch_bounds__(32)
void k_gemm_wmma(const _Float16* __restrict__ A, const _Float16* __restrict__ W,
                 const float* __restrict__ bias, float* __restrict__ outF,
                 _Float16* __restrict__ outH, int N, int K, int act) {
  const int lane = threadIdx.x;
  const int l15 = lane & 15;
  const int m0 = blockIdx.x * 32, n0 = blockIdx.y * 32;
  const int akb = (lane < 16) ? 0 : 8;
  const int bkb = (lane < 16) ? 0 : 8;
  const int arow0 = m0 + l15;
  const int ncol0 = n0 + l15, ncol1 = ncol0 + 16;
  const bool bval0 = ncol0 < N, bval1 = ncol1 < N;

  const _Float16* pa0 = A + (size_t)arow0 * K + akb;
  const _Float16* pa1 = pa0 + (size_t)16 * K;
  const _Float16* pb0 = W + (size_t)(bval0 ? ncol0 : 0) * K + bkb;
  const _Float16* pb1 = W + (size_t)(bval1 ? ncol1 : 0) * K + bkb;
  const v8h zf = {};

  v8f c00 = {}, c01 = {}, c10 = {}, c11 = {};
  for (int k0 = 0; k0 < K; k0 += 32) {
    v16h a0 = pack16(*(const v8h*)(pa0), *(const v8h*)(pa0 + 16));
    v16h a1 = pack16(*(const v8h*)(pa1), *(const v8h*)(pa1 + 16));
    v8h b0lo = *(const v8h*)(pb0), b0hi = *(const v8h*)(pb0 + 16);
    v8h b1lo = *(const v8h*)(pb1), b1hi = *(const v8h*)(pb1 + 16);
    pa0 += 32; pa1 += 32; pb0 += 32; pb1 += 32;
    __builtin_prefetch(pb0 + 480, 0, 1);   // stream the weight matrix
    if (!bval0) { b0lo = zf; b0hi = zf; }
    if (!bval1) { b1lo = zf; b1hi = zf; }
    v16h b0 = pack16(b0lo, b0hi), b1 = pack16(b1lo, b1hi);
    c00 = wmma_f16(a0, b0, c00);
    c01 = wmma_f16(a0, b1, c01);
    c10 = wmma_f16(a1, b0, c10);
    c11 = wmma_f16(a1, b1, c11);
  }

  const float bv0 = (bval0 && bias) ? bias[ncol0] : 0.f;
  const float bv1 = (bval1 && bias) ? bias[ncol1] : 0.f;
  const int mb = (lane < 16) ? 0 : 8;
#pragma unroll
  for (int r = 0; r < 8; ++r) {
    int ma = m0 + mb + r, mc = ma + 16;
    if (bval0) {
      float v = act_apply(c00[r] + bv0, act);
      float u = act_apply(c10[r] + bv0, act);
      if (outF) { outF[(size_t)ma * N + ncol0] = v; outF[(size_t)mc * N + ncol0] = u; }
      if (outH) { outH[(size_t)ma * N + ncol0] = (_Float16)v; outH[(size_t)mc * N + ncol0] = (_Float16)u; }
    }
    if (bval1) {
      float v = act_apply(c01[r] + bv1, act);
      float u = act_apply(c11[r] + bv1, act);
      if (outF) { outF[(size_t)ma * N + ncol1] = v; outF[(size_t)mc * N + ncol1] = u; }
      if (outH) { outH[(size_t)ma * N + ncol1] = (_Float16)v; outH[(size_t)mc * N + ncol1] = (_Float16)u; }
    }
  }
}

// ---------------------------------------------------------------------------
// NHWC im2col fragment gathers (branch-free clamp + register select)
// ---------------------------------------------------------------------------

__device__ __forceinline__ v8h conv_gather(const _Float16* __restrict__ X, int bb,
                                           int oh, int ow, int H, int Wd, int cl,
                                           int Cin, int ks) {
  int cell = ks >> cl, cs = ks & (Cin - 1);
  int kh = cell >> 2, kw = cell & 3;
  int ih = (oh << 1) - 1 + kh, iw = (ow << 1) - 1 + kw;
  bool v = (ih >= 0) & (ih < H) & (iw >= 0) & (iw < Wd);
  size_t base = v ? (((((size_t)bb * H + ih) * Wd + iw) << cl) + cs) : 0;
  v8h r = *(const v8h*)(X + base);
  v8h zf = {};
  if (!v) r = zf;
  return r;
}

__device__ __forceinline__ v8h deconv_gather(const _Float16* __restrict__ X, int bb,
                                             int oh, int ow, int H, int Wd, int cl,
                                             int Cin, int ks) {
  int cell = ks >> cl, cs = ks & (Cin - 1);
  int kh = cell >> 2, kw = cell & 3;
  int ihd = oh + kh - 2, iwd = ow + kw - 2;
  int ih = ihd >> 1, iw = iwd >> 1;
  bool v = (ihd >= 0) & (iwd >= 0) & !(ihd & 1) & !(iwd & 1) & (ih < H) & (iw < Wd);
  size_t base = v ? (((((size_t)bb * H + ih) * Wd + iw) << cl) + cs) : 0;
  v8h r = *(const v8h*)(X + base);
  v8h zf = {};
  if (!v) r = zf;
  return r;
}

// ---------------------------------------------------------------------------
// NHWC implicit-GEMM conv: 4x4, stride 2, pad 1, ReLU. 32x32 tile per wave.
// X: [B,H,W,Cin] f16 (Cin = 1<<cl >= 8). Wr: [Cout][(kh*4+kw)*Cin+ci].
// Y NHWC flat index = m*Cout + n. Requires M % 32 == 0.
// ---------------------------------------------------------------------------

__global__ __launch_bounds__(32)
void k_conv4x4s2_wmma(const _Float16* __restrict__ X, const _Float16* __restrict__ Wr,
                      const float* __restrict__ bias, _Float16* __restrict__ Y,
                      int H, int Wd, int cl, int Cout, int ohwBits, int owBits) {
  const int Cin = 1 << cl;
  const int K = Cin << 4;
  const int lane = threadIdx.x;
  const int l15 = lane & 15;
  const int m0 = blockIdx.x * 32;
  const int n0 = blockIdx.y * 32;
  const int akb = (lane < 16) ? 0 : 8;
  const int bkb = (lane < 16) ? 0 : 8;

  const int arow0 = m0 + l15, arow1 = arow0 + 16;
  const int bb0 = arow0 >> ohwBits;
  const int rem0 = arow0 & ((1 << ohwBits) - 1);
  const int oh0 = rem0 >> owBits, ow0 = rem0 & ((1 << owBits) - 1);
  const int bb1 = arow1 >> ohwBits;
  const int rem1 = arow1 & ((1 << ohwBits) - 1);
  const int oh1 = rem1 >> owBits, ow1 = rem1 & ((1 << owBits) - 1);

  const int ncol0 = n0 + l15, ncol1 = ncol0 + 16;
  const bool bval0 = ncol0 < Cout, bval1 = ncol1 < Cout;
  const _Float16* pb0 = Wr + (size_t)(bval0 ? ncol0 : 0) * K + bkb;
  const _Float16* pb1 = Wr + (size_t)(bval1 ? ncol1 : 0) * K + bkb;
  const v8h zf = {};

  v8f c00 = {}, c01 = {}, c10 = {}, c11 = {};
  for (int k0 = 0; k0 < K; k0 += 32) {
    int ks = k0 + akb;
    v16h a0 = pack16(conv_gather(X, bb0, oh0, ow0, H, Wd, cl, Cin, ks),
                     conv_gather(X, bb0, oh0, ow0, H, Wd, cl, Cin, ks + 16));
    v16h a1 = pack16(conv_gather(X, bb1, oh1, ow1, H, Wd, cl, Cin, ks),
                     conv_gather(X, bb1, oh1, ow1, H, Wd, cl, Cin, ks + 16));
    v8h b0lo = *(const v8h*)(pb0), b0hi = *(const v8h*)(pb0 + 16);
    v8h b1lo = *(const v8h*)(pb1), b1hi = *(const v8h*)(pb1 + 16);
    pb0 += 32; pb1 += 32;
    if (!bval0) { b0lo = zf; b0hi = zf; }
    if (!bval1) { b1lo = zf; b1hi = zf; }
    v16h b0 = pack16(b0lo, b0hi), b1 = pack16(b1lo, b1hi);
    c00 = wmma_f16(a0, b0, c00);
    c01 = wmma_f16(a0, b1, c01);
    c10 = wmma_f16(a1, b0, c10);
    c11 = wmma_f16(a1, b1, c11);
  }

  const float bv0 = (bval0 && bias) ? bias[ncol0] : 0.f;
  const float bv1 = (bval1 && bias) ? bias[ncol1] : 0.f;
  const int mb = (lane < 16) ? 0 : 8;
#pragma unroll
  for (int r = 0; r < 8; ++r) {
    int ma = m0 + mb + r, mc = ma + 16;
    if (bval0) {
      float v = c00[r] + bv0; v = v > 0.f ? v : 0.f;
      float u = c10[r] + bv0; u = u > 0.f ? u : 0.f;
      Y[(size_t)ma * Cout + ncol0] = (_Float16)v;
      Y[(size_t)mc * Cout + ncol0] = (_Float16)u;
    }
    if (bval1) {
      float v = c01[r] + bv1; v = v > 0.f ? v : 0.f;
      float u = c11[r] + bv1; u = u > 0.f ? u : 0.f;
      Y[(size_t)ma * Cout + ncol1] = (_Float16)v;
      Y[(size_t)mc * Cout + ncol1] = (_Float16)u;
    }
  }
}

// ---------------------------------------------------------------------------
// NHWC implicit-GEMM transposed conv (lhs_dilation 2, pad 2, OH=2H),
// 32x32 tile per wave. Wr has flip/transpose baked in.
// ---------------------------------------------------------------------------

__global__ __launch_bounds__(32)
void k_deconv4x4_wmma(const _Float16* __restrict__ X, const _Float16* __restrict__ Wr,
                      const float* __restrict__ bias, _Float16* __restrict__ YH,
                      float* __restrict__ YF, int H, int Wd, int cl, int Cout,
                      int ohwBits, int owBits, int relu) {
  const int Cin = 1 << cl;
  const int K = Cin << 4;
  const int lane = threadIdx.x;
  const int l15 = lane & 15;
  const int m0 = blockIdx.x * 32;
  const int n0 = blockIdx.y * 32;
  const int akb = (lane < 16) ? 0 : 8;
  const int bkb = (lane < 16) ? 0 : 8;

  const int arow0 = m0 + l15, arow1 = arow0 + 16;
  const int bb0 = arow0 >> ohwBits;
  const int rem0 = arow0 & ((1 << ohwBits) - 1);
  const int oh0 = rem0 >> owBits, ow0 = rem0 & ((1 << owBits) - 1);
  const int bb1 = arow1 >> ohwBits;
  const int rem1 = arow1 & ((1 << ohwBits) - 1);
  const int oh1 = rem1 >> owBits, ow1 = rem1 & ((1 << owBits) - 1);

  const int ncol0 = n0 + l15, ncol1 = ncol0 + 16;
  const bool bval0 = ncol0 < Cout, bval1 = ncol1 < Cout;
  const _Float16* pb0 = Wr + (size_t)(bval0 ? ncol0 : 0) * K + bkb;
  const _Float16* pb1 = Wr + (size_t)(bval1 ? ncol1 : 0) * K + bkb;
  const v8h zf = {};

  v8f c00 = {}, c01 = {}, c10 = {}, c11 = {};
  for (int k0 = 0; k0 < K; k0 += 32) {
    int ks = k0 + akb;
    v16h a0 = pack16(deconv_gather(X, bb0, oh0, ow0, H, Wd, cl, Cin, ks),
                     deconv_gather(X, bb0, oh0, ow0, H, Wd, cl, Cin, ks + 16));
    v16h a1 = pack16(deconv_gather(X, bb1, oh1, ow1, H, Wd, cl, Cin, ks),
                     deconv_gather(X, bb1, oh1, ow1, H, Wd, cl, Cin, ks + 16));
    v8h b0lo = *(const v8h*)(pb0), b0hi = *(const v8h*)(pb0 + 16);
    v8h b1lo = *(const v8h*)(pb1), b1hi = *(const v8h*)(pb1 + 16);
    pb0 += 32; pb1 += 32;
    if (!bval0) { b0lo = zf; b0hi = zf; }
    if (!bval1) { b1lo = zf; b1hi = zf; }
    v16h b0 = pack16(b0lo, b0hi), b1 = pack16(b1lo, b1hi);
    c00 = wmma_f16(a0, b0, c00);
    c01 = wmma_f16(a0, b1, c01);
    c10 = wmma_f16(a1, b0, c10);
    c11 = wmma_f16(a1, b1, c11);
  }

  const float bv0 = (bval0 && bias) ? bias[ncol0] : 0.f;
  const float bv1 = (bval1 && bias) ? bias[ncol1] : 0.f;
  const int mb = (lane < 16) ? 0 : 8;
#pragma unroll
  for (int r = 0; r < 8; ++r) {
    int ma = m0 + mb + r, mc = ma + 16;
    if (bval0) {
      float v = c00[r] + bv0;
      float u = c10[r] + bv0;
      if (relu) { v = v > 0.f ? v : 0.f; u = u > 0.f ? u : 0.f; }
      size_t ia = (size_t)ma * Cout + ncol0, ic = (size_t)mc * Cout + ncol0;
      if (YH) { YH[ia] = (_Float16)v; YH[ic] = (_Float16)u; }
      if (YF) { YF[ia] = v; YF[ic] = u; }
    }
    if (bval1) {
      float v = c01[r] + bv1;
      float u = c11[r] + bv1;
      if (relu) { v = v > 0.f ? v : 0.f; u = u > 0.f ? u : 0.f; }
      size_t ia = (size_t)ma * Cout + ncol1, ic = (size_t)mc * Cout + ncol1;
      if (YH) { YH[ia] = (_Float16)v; YH[ic] = (_Float16)u; }
      if (YF) { YF[ia] = v; YF[ic] = u; }
    }
  }
}

// ---------------------------------------------------------------------------
// Host orchestration
// ---------------------------------------------------------------------------

extern "C" void kernel_launch(void* const* d_in, const int* in_sizes, int n_in,
                              void* d_out, int out_size, void* d_ws, size_t ws_size,
                              hipStream_t stream) {
  (void)in_sizes; (void)n_in; (void)out_size; (void)ws_size;

  const int LATENT = 256, TARGET = 128, HIDn = 512, DS = 16;
  const int KPAD = 288;  // 262 padded to a multiple of 32

  const float* img      = (const float*)d_in[0];
  const float* action   = (const float*)d_in[1];
  const float* z_prev   = (const float*)d_in[2];
  const float* ssm_prev = (const float*)d_in[3];
  const float* eps      = (const float*)d_in[4];
  const float* conv_w[4] = {(const float*)d_in[5], (const float*)d_in[7],
                            (const float*)d_in[9], (const float*)d_in[11]};
  const float* conv_b[4] = {(const float*)d_in[6], (const float*)d_in[8],
                            (const float*)d_in[10], (const float*)d_in[12]};
  const float* enc_fc_w = (const float*)d_in[13];
  const float* enc_fc_b = (const float*)d_in[14];
  const float* post_w   = (const float*)d_in[15];
  const float* post_b   = (const float*)d_in[16];
  const float* prior_w  = (const float*)d_in[17];
  const float* prior_b  = (const float*)d_in[18];
  const float* ssm_in_w = (const float*)d_in[19];
  const float* ssm_in_b = (const float*)d_in[20];
  const float* A_log    = (const float*)d_in[21];
  const float* B_w      = (const float*)d_in[22];
  const float* C_w      = (const float*)d_in[23];
  const float* Delta_w  = (const float*)d_in[24];
  const float* Delta_b  = (const float*)d_in[25];
  const float* Dv       = (const float*)d_in[26];
  const float* dec_fc_w = (const float*)d_in[27];
  const float* dec_fc_b = (const float*)d_in[28];
  const float* dc_w[4]  = {(const float*)d_in[29], (const float*)d_in[31],
                           (const float*)d_in[33], (const float*)d_in[35]};
  const float* dc_b[4]  = {(const float*)d_in[30], (const float*)d_in[32],
                           (const float*)d_in[34], (const float*)d_in[36]};

  float* out = (float*)d_out;
  float* out_logits = out;                    // [64,1,256,256] (NHWC C=1 == NCHW)
  float* out_z      = out + 4194304;          // [64,256]
  float* out_state  = out_z + 16384;          // [64,512,16]
  float* out_mupri  = out_state + 524288;     // [64,256]
  float* out_lvpri  = out_mupri + 16384;
  float* out_mupost = out_lvpri + 16384;
  float* out_lvpost = out_mupost + 16384;

  char*  wsp = (char*)d_ws;
  size_t off = 0;
  auto alloc = [&](size_t bytes) -> void* {
    size_t cur = off;
    off = (cur + bytes + 255) & ~(size_t)255;
    return (void*)(wsp + cur);
  };
  auto allocH = [&](long n) -> _Float16* { return (_Float16*)alloc((size_t)n * 2); };
  auto allocF = [&](long n) -> float*    { return (float*)alloc((size_t)n * 4); };

  // f16 weights (reordered) -------------------------------------------------
  _Float16* img8    = allocH(64L * 256 * 256 * 8);       // NHWC, C padded to 8
  _Float16* cw16[4] = {allocH(32 * 16 * 8), allocH(64 * 16 * 32),
                       allocH(128 * 16 * 64), allocH(256 * 16 * 128)};
  _Float16* encw16  = allocH(512L * 65536);              // NHWC-k permuted
  _Float16* postw16 = allocH(512 * 1024);
  _Float16* priw16  = allocH(512 * 512);
  _Float16* ssmw16  = allocH(512 * KPAD);                // K padded 262->288
  _Float16* bw16    = allocH(16 * 512);
  _Float16* cwv16   = allocH(16 * 512);
  _Float16* dlw16   = allocH(512 * 512);
  _Float16* decw16  = allocH(32768L * 128);              // rows permuted to NHWC
  float*    decb_r  = allocF(32768);                     // bias permuted to match
  _Float16* dcw16[4]= {allocH(64 * 16 * 128), allocH(32 * 16 * 64),
                       allocH(16 * 16 * 32), allocH(1 * 16 * 16)};
  // activations (NHWC) ------------------------------------------------------
  _Float16* zcat16  = allocH(64 * KPAD);
  _Float16* ssmin16 = allocH(64 * HIDn);
  float*    ssminF  = allocF(64 * HIDn);
  float*    btF     = allocF(64 * DS);
  float*    ctF     = allocF(64 * DS);
  float*    deltaF  = allocF(64 * HIDn);
  float*    belF    = allocF(64 * HIDn);
  _Float16* bel16   = allocH(64 * HIDn);
  float*    priorF  = allocF(64 * 2 * LATENT);
  _Float16* h1      = allocH(64L * 128 * 128 * 32);
  _Float16* h2      = allocH(64L * 64 * 64 * 64);
  _Float16* h3      = allocH(64L * 32 * 32 * 128);
  _Float16* h4      = allocH(64L * 16 * 16 * 256);
  _Float16* feat16  = allocH(64 * 512);
  _Float16* pcat16  = allocH(64 * 1024);
  float*    postF   = allocF(64 * 2 * LATENT);
  _Float16* ztgt16  = allocH(64 * TARGET);
  _Float16* x0      = allocH(64L * 16 * 16 * 128);
  _Float16* dA      = allocH(64L * 32 * 32 * 64);
  _Float16* dB      = allocH(64L * 64 * 64 * 32);
  _Float16* dC      = allocH(64L * 128 * 128 * 16);

  auto g1 = [&](long n) { return dim3((unsigned)((n + 255) / 256)); };
  auto cvt = [&](const float* s, _Float16* d, long n) {
    k_f32_to_f16<<<g1(n), dim3(256), 0, stream>>>(s, d, n);
  };
  auto gemm = [&](const _Float16* A, const _Float16* W, const float* bias,
                  float* oF, _Float16* oH, int M, int N, int K, int act) {
    dim3 g(M / 32, (N + 31) / 32);
    k_gemm_wmma<<<g, dim3(32), 0, stream>>>(A, W, bias, oF, oH, N, K, act);
  };

  // --- one-time weight/image preparation -----------------------------------
  k_img8<<<g1(64L * 256 * 256 * 8), dim3(256), 0, stream>>>(img, img8);
  k_wr_conv<<<g1(32 * 16 * 8),    dim3(256), 0, stream>>>(conv_w[0], cw16[0], 32, 3, 8);
  k_wr_conv<<<g1(64 * 16 * 32),   dim3(256), 0, stream>>>(conv_w[1], cw16[1], 64, 32, 32);
  k_wr_conv<<<g1(128 * 16 * 64),  dim3(256), 0, stream>>>(conv_w[2], cw16[2], 128, 64, 64);
  k_wr_conv<<<g1(256 * 16 * 128), dim3(256), 0, stream>>>(conv_w[3], cw16[3], 256, 128, 128);
  k_wr_enc<<<g1(512L * 65536), dim3(256), 0, stream>>>(enc_fc_w, encw16);
  cvt(post_w, postw16, 512 * 1024);
  cvt(prior_w, priw16, 512 * 512);
  k_wr_pad<<<g1(512 * KPAD), dim3(256), 0, stream>>>(ssm_in_w, ssmw16, 512, 262, KPAD);
  cvt(B_w, bw16, 16 * 512);
  cvt(C_w, cwv16, 16 * 512);
  cvt(Delta_w, dlw16, 512 * 512);
  k_wr_dec<<<g1(32768L * 128), dim3(256), 0, stream>>>(dec_fc_w, decw16);
  k_wr_decb<<<g1(32768), dim3(256), 0, stream>>>(dec_fc_b, decb_r);
  k_wr_deconv<<<g1(64 * 16 * 128), dim3(256), 0, stream>>>(dc_w[0], dcw16[0], 128, 64);
  k_wr_deconv<<<g1(32 * 16 * 64),  dim3(256), 0, stream>>>(dc_w[1], dcw16[1], 64, 32);
  k_wr_deconv<<<g1(16 * 16 * 32),  dim3(256), 0, stream>>>(dc_w[2], dcw16[2], 32, 16);
  k_wr_deconv<<<g1(1 * 16 * 16),   dim3(256), 0, stream>>>(dc_w[3], dcw16[3], 16, 1);

  // --- SSM branch ----------------------------------------------------------
  k_zcat<<<g1(64 * KPAD), dim3(256), 0, stream>>>(z_prev, action, zcat16, KPAD);
  gemm(zcat16, ssmw16, ssm_in_b, ssminF, ssmin16, 64, HIDn, KPAD, ACT_SILU);
  gemm(ssmin16, bw16, nullptr, btF, nullptr, 64, DS, HIDn, ACT_NONE);
  gemm(ssmin16, cwv16, nullptr, ctF, nullptr, 64, DS, HIDn, ACT_NONE);
  gemm(ssmin16, dlw16, Delta_b, deltaF, nullptr, 64, HIDn, HIDn, ACT_SOFTPLUS);
  k_ssm<<<g1(64 * HIDn), dim3(256), 0, stream>>>(ssminF, deltaF, btF, ctF, A_log,
                                                 Dv, ssm_prev, out_state, belF,
                                                 bel16, 64, HIDn, DS);
  gemm(bel16, priw16, prior_b, priorF, nullptr, 64, 2 * LATENT, HIDn, ACT_NONE);
  k_split_halves<<<g1(64 * LATENT), dim3(256), 0, stream>>>(priorF, 64, LATENT,
                                                            out_mupri, out_lvpri);

  // --- CNN encoder (NHWC implicit-GEMM WMMA) -------------------------------
  auto conv = [&](const _Float16* X, int H, int cl, int Cout, int wi, _Float16* Y,
                  int ohwBits, int owBits) {
    int M = 64 << ohwBits;
    dim3 g(M / 32, (Cout + 31) / 32);
    k_conv4x4s2_wmma<<<g, dim3(32), 0, stream>>>(X, cw16[wi], conv_b[wi], Y, H, H,
                                                 cl, Cout, ohwBits, owBits);
  };
  conv(img8, 256, 3, 32, 0, h1, 14, 7);   // 256x256x8  -> 128x128x32
  conv(h1,   128, 5, 64, 1, h2, 12, 6);   // 128x128x32 -> 64x64x64
  conv(h2,    64, 6, 128, 2, h3, 10, 5);  // 64x64x64   -> 32x32x128
  conv(h3,    32, 7, 256, 3, h4, 6, 3);   // 32x32x128  -> 8x8x256
  // h4 NHWC row-major [64, 65536] matches the permuted enc_fc weight
  gemm(h4, encw16, enc_fc_b, nullptr, feat16, 64, 512, 65536, ACT_RELU);

  // --- posterior + reparameterization --------------------------------------
  k_concat2_h<<<g1(64 * 1024), dim3(256), 0, stream>>>(feat16, 512, bel16, 512,
                                                       pcat16, 64);
  gemm(pcat16, postw16, post_b, postF, nullptr, 64, 2 * LATENT, 1024, ACT_NONE);
  k_reparam<<<g1(64 * LATENT), dim3(256), 0, stream>>>(postF, eps, out_z,
                                                       out_mupost, out_lvpost,
                                                       ztgt16, 64, LATENT, TARGET);

  // --- decoder -------------------------------------------------------------
  // dec_fc with permuted rows directly produces NHWC x0 [64,16,16,128]
  gemm(ztgt16, decw16, decb_r, nullptr, x0, 64, 32768, TARGET, ACT_NONE);
  auto deconv = [&](const _Float16* X, int H, int cl, int Cout, int wi,
                    _Float16* YH, float* YF, int relu, int ohwBits, int owBits) {
    int M = 64 << ohwBits;
    dim3 g(M / 32, (Cout + 31) / 32);
    k_deconv4x4_wmma<<<g, dim3(32), 0, stream>>>(X, dcw16[wi], dc_b[wi], YH, YF,
                                                 H, H, cl, Cout, ohwBits, owBits,
                                                 relu);
  };
  deconv(x0, 16, 7, 64, 0, dA, nullptr, 1, 10, 5);     // 16->32, C 128->64
  deconv(dA, 32, 6, 32, 1, dB, nullptr, 1, 12, 6);     // 32->64, C 64->32
  deconv(dB, 64, 5, 16, 2, dC, nullptr, 1, 14, 7);     // 64->128, C 32->16
  deconv(dC, 128, 4, 1, 3, nullptr, out_logits, 0, 16, 8); // 128->256, C 16->1
}
